// MoEHead_2894807957601
// MI455X (gfx1250) — matrix operand order repaired and checked by reading, not compiled, not run on hardware
//
#include <hip/hip_runtime.h>
#include <math.h>

// ---------------- problem constants ----------------
#define N_TOK  32768
#define DIM    1024
#define HID    1024
#define NEXP   8
#define TOPK   2
#define TAU    1.2f
#define LN_EPS 1e-5f

typedef __attribute__((ext_vector_type(16))) __bf16 v16bf;
typedef __attribute__((ext_vector_type(8)))  float  v8f;

// ---------------- LDS tiling ----------------
#define TILE_ROWS     64
#define LDS_STRIDE_E  1032                       // bf16 elems per row (1024 + 8 pad -> conflict-free A frags)
#define LDS_STRIDE_B  (LDS_STRIDE_E * 2)         // 2064 bytes
#define LDS_TILE_BYTES (TILE_ROWS * LDS_STRIDE_B) // 132096
#define SMEM_BYTES    (2 * LDS_TILE_BYTES)        // 264192  (fits 320KB WGP LDS)

// ---------------- workspace layout (bytes) ----------------
#define OFF_ZB        ((size_t)0)                      // [N][D] bf16          64 MB
#define OFF_W1T       ((size_t)67108864)               // [E][H][D] bf16       16 MB (transposed)
#define OFF_W2T       ((size_t)83886080)               // [E][H][H] bf16       16 MB (transposed)
#define OFF_SLOT_TOK  ((size_t)100663296)              // [E][N] int            1 MB
#define OFF_YSLOT     ((size_t)101711872)              // [E][N] float          1 MB
#define OFF_CNT       ((size_t)102760448)              // [E] int (+pad)
#define OFF_TK_E      ((size_t)102760512)              // [N][K] int
#define OFF_TK_SLOT   ((size_t)103022656)              // [N][K] int
#define OFF_TK_W      ((size_t)103284800)              // [N][K] float
// total ~103.6 MB

// ---------------- helpers ----------------
__device__ __forceinline__ v8f zero8() {
  v8f v;
  #pragma unroll
  for (int i = 0; i < 8; ++i) v[i] = 0.0f;
  return v;
}

__device__ __forceinline__ v16bf frag_from2(uint4 a, uint4 b) {
  union { uint4 q[2]; v16bf v; } u;
  u.q[0] = a; u.q[1] = b;
  return u.v;
}

// A fragment (16x32 bf16) from LDS, rows rowBase..rowBase+15, K = k..k+31.
// ISA layout: lanes 0-15 row=lane hold K [k..k+8) and [k+16..k+24);
//             lanes 16-31 row=lane-16 hold K [k+8..k+16) and [k+24..k+32).
__device__ __forceinline__ v16bf load_a_lds(const char* tile, int rowBase, int k, unsigned lane) {
  int row = rowBase + (int)(lane & 15u);
  int kb  = k + ((lane & 16u) ? 8 : 0);
  const char* p = tile + row * LDS_STRIDE_B + kb * 2;
  uint4 lo = *(const uint4*)p;
  uint4 hi = *(const uint4*)(p + 32);
  return frag_from2(lo, hi);
}

// B fragment (32x16 bf16) from global weights stored [n][k] (k contiguous).
// ISA layout: lanes 0-15 col=lane hold K [k..k+8) and [k+16..k+24); lanes 16-31 col=lane-16 hold K [k+8..k+16) and [k+24..k+32).
__device__ __forceinline__ v16bf load_b_glob(const __bf16* w, int n, int k, unsigned lane) {
  int col = n + (int)(lane & 15u);
  int kb  = k + ((lane & 16u) ? 8 : 0);
  const char* p = (const char*)(w + (size_t)col * DIM + kb);
  uint4 lo = *(const uint4*)p;
  uint4 hi = *(const uint4*)(p + 32);
  return frag_from2(lo, hi);
}

__device__ __forceinline__ float gelu_exact(float x) {
  return 0.5f * x * (1.0f + erff(x * 0.70710678118654752f));
}

// ---------------- prep kernels ----------------
__global__ void zero_cnt_kernel(int* cnt) {
  if (threadIdx.x < NEXP) cnt[threadIdx.x] = 0;
}

__global__ void cvt_z_kernel(const float* __restrict__ z, __bf16* __restrict__ zb) {
  size_t i = (size_t)blockIdx.x * blockDim.x + threadIdx.x;
  if (i < (size_t)N_TOK * DIM) zb[i] = (__bf16)z[i];
}

// transpose+convert: wt[e][n][k] = w[e][k][n]  (square 1024x1024 per expert)
__global__ void cvt_wt_kernel(const float* __restrict__ w, __bf16* __restrict__ wt) {
  size_t i = (size_t)blockIdx.x * blockDim.x + threadIdx.x;
  if (i >= (size_t)NEXP * DIM * HID) return;
  int e   = (int)(i / ((size_t)DIM * HID));
  int rem = (int)(i % ((size_t)DIM * HID));
  int n = rem / DIM;
  int k = rem % DIM;
  wt[i] = (__bf16)w[(size_t)e * DIM * HID + (size_t)k * HID + n];
}

// ---------------- router ----------------
__global__ void router_kernel(const float* __restrict__ z, const float* __restrict__ Wr,
                              const float* __restrict__ br,
                              int* cnt, int* slot_tok,
                              int* tk_e, int* tk_slot, float* tk_w,
                              float* out_probs, float* out_topk) {
  unsigned lane = threadIdx.x & 31u;
  unsigned wid  = threadIdx.x >> 5;
  int t = blockIdx.x * 8 + (int)wid;   // one wave per token, exact cover

  float acc[NEXP];
  #pragma unroll
  for (int e = 0; e < NEXP; ++e) acc[e] = 0.0f;
  const float* zr = z + (size_t)t * DIM;
  for (int d = (int)lane; d < DIM; d += 32) {
    float zv = zr[d];
    const float* wr = Wr + (size_t)d * NEXP;
    #pragma unroll
    for (int e = 0; e < NEXP; ++e) acc[e] += zv * wr[e];
  }
  #pragma unroll
  for (int off = 16; off > 0; off >>= 1) {
    #pragma unroll
    for (int e = 0; e < NEXP; ++e) acc[e] += __shfl_xor(acc[e], off, 32);
  }

  if (lane == 0) {
    float p[NEXP];
    float mx = -1e30f;
    #pragma unroll
    for (int e = 0; e < NEXP; ++e) { acc[e] += br[e]; mx = fmaxf(mx, acc[e]); }
    float sum = 0.0f;
    #pragma unroll
    for (int e = 0; e < NEXP; ++e) { p[e] = expf((acc[e] - mx) * (1.0f / TAU)); sum += p[e]; }
    float inv = 1.0f / sum;
    #pragma unroll
    for (int e = 0; e < NEXP; ++e) { p[e] *= inv; out_probs[(size_t)t * NEXP + e] = p[e]; }

    // top-2, ties -> lowest index (matches jax top_k)
    int e0 = 0;
    #pragma unroll
    for (int e = 1; e < NEXP; ++e) if (p[e] > p[e0]) e0 = e;
    int e1 = (e0 == 0) ? 1 : 0;
    #pragma unroll
    for (int e = 0; e < NEXP; ++e) if (e != e0 && p[e] > p[e1]) e1 = e;

    float denom = p[e0] + p[e1];
    denom = fmaxf(denom, 1e-8f);

    int i0 = atomicAdd(&cnt[e0], 1);
    slot_tok[e0 * N_TOK + i0] = t;
    tk_e[2 * t] = e0; tk_slot[2 * t] = i0; tk_w[2 * t] = p[e0] / denom;

    int i1 = atomicAdd(&cnt[e1], 1);
    slot_tok[e1 * N_TOK + i1] = t;
    tk_e[2 * t + 1] = e1; tk_slot[2 * t + 1] = i1; tk_w[2 * t + 1] = p[e1] / denom;

    out_topk[2 * t]     = (float)e0;
    out_topk[2 * t + 1] = (float)e1;
  }
}

// ---------------- fused expert block: GEMM1+GELU -> GEMM2+residual -> LN -> head ----------------
__global__ __launch_bounds__(256) void expert_kernel(
    const __bf16* __restrict__ zb, const __bf16* __restrict__ w1t, const __bf16* __restrict__ w2t,
    const float* __restrict__ b1, const float* __restrict__ b2,
    const float* __restrict__ gamma, const float* __restrict__ beta,
    const float* __restrict__ Wo, const float* __restrict__ bo,
    const int* __restrict__ cnt, const int* __restrict__ slot_tok,
    float* __restrict__ yslot) {
  extern __shared__ char smem[];
  char* zg = smem;                    // 64 x 1032 bf16: gathered Z tile, later (H2 + residual)
  char* h1 = smem + LDS_TILE_BYTES;   // 64 x 1032 bf16: gelu(Z@W1+b1)

  const int e  = blockIdx.x / (N_TOK / TILE_ROWS);
  const int mb = blockIdx.x % (N_TOK / TILE_ROWS);
  const int cn = cnt[e];
  if (mb * TILE_ROWS >= cn) return;

  const unsigned tid  = threadIdx.x;
  const unsigned lane = tid & 31u;
  const unsigned wid  = tid >> 5;     // 8 waves
  const int wm = (int)(wid & 3u);     // row strip: rows wm*16 .. wm*16+15
  const int wn = (int)(wid >> 2);     // col half within 256-col chunk

  // ---- stage gathered Z tile into LDS (4 threads per row, 16B vector copies) ----
  {
    int r    = (int)(tid >> 2);
    int cq   = (int)(tid & 3u) * 16;
    int slot = mb * TILE_ROWS + r;
    int tok  = (slot < cn) ? slot_tok[e * N_TOK + slot] : slot_tok[e * N_TOK];
    const uint4* src = (const uint4*)(zb + (size_t)tok * DIM);
    uint4* dst = (uint4*)(zg + r * LDS_STRIDE_B);
    #pragma unroll
    for (int q = 0; q < 16; ++q) dst[cq + q] = src[cq + q];
  }
  __syncthreads();

  const __bf16* w1e = w1t + (size_t)e * HID * DIM;
  const __bf16* w2e = w2t + (size_t)e * HID * HID;

  // ---- GEMM1: h1 = gelu(Zg @ W1 + b1), weights stream from L2 ----
  for (int nb0 = 0; nb0 < HID; nb0 += 256) {
    const int nb = nb0 + wn * 128;
    v8f acc[8];
    #pragma unroll
    for (int t = 0; t < 8; ++t) acc[t] = zero8();
    for (int k = 0; k < DIM; k += 32) {
      __builtin_prefetch((const void*)(w1e + (size_t)(nb + (int)(lane & 15u)) * DIM + ((k + 256) & (DIM - 1))), 0, 0);
      v16bf a = load_a_lds(zg, wm * 16, k, lane);
      #pragma unroll
      for (int t = 0; t < 8; ++t) {
        v16bf b = load_b_glob(w1e, nb + t * 16, k, lane);
        acc[t] = __builtin_amdgcn_wmma_f32_16x16x32_bf16(false, a, false, b, (short)0, acc[t], false, false);
      }
    }
    #pragma unroll
    for (int t = 0; t < 8; ++t) {
      int col = nb + t * 16 + (int)(lane & 15u);
      float bias = b1[e * HID + col];
      #pragma unroll
      for (int j = 0; j < 8; ++j) {
        int rl = wm * 16 + j + ((lane & 16u) ? 8 : 0);
        float v = gelu_exact(acc[t][j] + bias);
        *(__bf16*)(h1 + rl * LDS_STRIDE_B + col * 2) = (__bf16)v;
      }
    }
  }
  __syncthreads();

  // ---- GEMM2: zg <- (h1 @ W2 + b2) + Zg   (in-place residual, element read-then-write by owner wave) ----
  for (int nb0 = 0; nb0 < HID; nb0 += 256) {
    const int nb = nb0 + wn * 128;
    v8f acc[8];
    #pragma unroll
    for (int t = 0; t < 8; ++t) acc[t] = zero8();
    for (int k = 0; k < HID; k += 32) {
      __builtin_prefetch((const void*)(w2e + (size_t)(nb + (int)(lane & 15u)) * HID + ((k + 256) & (HID - 1))), 0, 0);
      v16bf a = load_a_lds(h1, wm * 16, k, lane);
      #pragma unroll
      for (int t = 0; t < 8; ++t) {
        v16bf b = load_b_glob(w2e, nb + t * 16, k, lane);
        acc[t] = __builtin_amdgcn_wmma_f32_16x16x32_bf16(false, a, false, b, (short)0, acc[t], false, false);
      }
    }
    #pragma unroll
    for (int t = 0; t < 8; ++t) {
      int col = nb + t * 16 + (int)(lane & 15u);
      float bias = b2[e * HID + col];
      #pragma unroll
      for (int j = 0; j < 8; ++j) {
        int rl = wm * 16 + j + ((lane & 16u) ? 8 : 0);
        __bf16* zp = (__bf16*)(zg + rl * LDS_STRIDE_B + col * 2);
        float v = acc[t][j] + bias + (float)(*zp);
        *zp = (__bf16)v;
      }
    }
  }
  __syncthreads();

  // ---- LayerNorm + head (H -> 1): one wave per row, 8 rows per wave ----
  for (int rr = 0; rr < 8; ++rr) {
    int rl   = (int)wid * 8 + rr;
    int slot = mb * TILE_ROWS + rl;
    const __bf16* rowp = (const __bf16*)(zg + rl * LDS_STRIDE_B);
    float xs[32];
    float s = 0.0f, s2 = 0.0f;
    #pragma unroll
    for (int i = 0; i < 32; ++i) {
      float x = (float)rowp[lane + 32 * i];
      xs[i] = x; s += x; s2 += x * x;
    }
    #pragma unroll
    for (int off = 16; off > 0; off >>= 1) {
      s  += __shfl_xor(s,  off, 32);
      s2 += __shfl_xor(s2, off, 32);
    }
    float mu   = s * (1.0f / HID);
    float var  = s2 * (1.0f / HID) - mu * mu;
    float rstd = rsqrtf(var + LN_EPS);
    float acc = 0.0f;
    #pragma unroll
    for (int i = 0; i < 32; ++i) {
      int c = (int)lane + 32 * i;
      float r = (xs[i] - mu) * rstd * gamma[e * HID + c] + beta[e * HID + c];
      acc += r * Wo[e * HID + c];
    }
    #pragma unroll
    for (int off = 16; off > 0; off >>= 1) acc += __shfl_xor(acc, off, 32);
    if (lane == 0 && slot < cn) yslot[e * N_TOK + slot] = acc + bo[e];
  }
}

// ---------------- finalize: gate-combine (fixed order -> deterministic), stats ----------------
__global__ void yfinal_kernel(const int* __restrict__ tk_e, const int* __restrict__ tk_slot,
                              const float* __restrict__ tk_w, const float* __restrict__ yslot,
                              float* __restrict__ out_y) {
  int t = blockIdx.x * blockDim.x + threadIdx.x;
  if (t >= N_TOK) return;
  float y = tk_w[2 * t]     * yslot[tk_e[2 * t]     * N_TOK + tk_slot[2 * t]]
          + tk_w[2 * t + 1] * yslot[tk_e[2 * t + 1] * N_TOK + tk_slot[2 * t + 1]];
  out_y[t] = y;
}

__global__ void stats_kernel(const float* __restrict__ probs, const int* __restrict__ cnt,
                             float* out_imp, float* out_load) {
  unsigned lane = threadIdx.x & 31u;
  unsigned e    = threadIdx.x >> 5;   // 8 waves = 8 experts
  float s = 0.0f;
  for (int t = (int)lane; t < N_TOK; t += 32) s += probs[(size_t)t * NEXP + e];
  #pragma unroll
  for (int off = 16; off > 0; off >>= 1) s += __shfl_xor(s, off, 32);
  if (lane == 0) {
    out_imp[e]  = s * (1.0f / N_TOK);
    out_load[e] = (float)cnt[e] * (1.0f / N_TOK);
  }
}

// ---------------- host launcher ----------------
extern "C" void kernel_launch(void* const* d_in, const int* in_sizes, int n_in,
                              void* d_out, int out_size, void* d_ws, size_t ws_size,
                              hipStream_t stream) {
  (void)in_sizes; (void)n_in; (void)out_size; (void)ws_size;

  const float* z     = (const float*)d_in[0];
  const float* Wr    = (const float*)d_in[1];
  const float* br    = (const float*)d_in[2];
  const float* W1    = (const float*)d_in[3];
  const float* b1    = (const float*)d_in[4];
  const float* W2    = (const float*)d_in[5];
  const float* b2    = (const float*)d_in[6];
  const float* gamma = (const float*)d_in[7];
  const float* beta  = (const float*)d_in[8];
  const float* Wo    = (const float*)d_in[9];
  const float* bo    = (const float*)d_in[10];

  char* ws = (char*)d_ws;
  __bf16* zb       = (__bf16*)(ws + OFF_ZB);
  __bf16* w1t      = (__bf16*)(ws + OFF_W1T);
  __bf16* w2t      = (__bf16*)(ws + OFF_W2T);
  int*    slot_tok = (int*)   (ws + OFF_SLOT_TOK);
  float*  yslot    = (float*) (ws + OFF_YSLOT);
  int*    cnt      = (int*)   (ws + OFF_CNT);
  int*    tk_e     = (int*)   (ws + OFF_TK_E);
  int*    tk_slot  = (int*)   (ws + OFF_TK_SLOT);
  float*  tk_w     = (float*) (ws + OFF_TK_W);

  float* out_y     = (float*)d_out;                                   // [N]
  float* out_probs = out_y + N_TOK;                                   // [N][E]
  float* out_topk  = out_probs + (size_t)N_TOK * NEXP;                // [N][K] (as float)
  float* out_imp   = out_topk + (size_t)N_TOK * TOPK;                 // [E]
  float* out_load  = out_imp + NEXP;                                  // [E]

  (void)hipFuncSetAttribute((const void*)expert_kernel,
                            hipFuncAttributeMaxDynamicSharedMemorySize, SMEM_BYTES);

  zero_cnt_kernel<<<1, 32, 0, stream>>>(cnt);
  cvt_z_kernel<<<(N_TOK * DIM + 255) / 256, 256, 0, stream>>>(z, zb);
  cvt_wt_kernel<<<(NEXP * DIM * HID + 255) / 256, 256, 0, stream>>>(W1, w1t);
  cvt_wt_kernel<<<(NEXP * DIM * HID + 255) / 256, 256, 0, stream>>>(W2, w2t);

  router_kernel<<<N_TOK / 8, 256, 0, stream>>>(z, Wr, br, cnt, slot_tok,
                                               tk_e, tk_slot, tk_w,
                                               out_probs, out_topk);

  expert_kernel<<<NEXP * (N_TOK / TILE_ROWS), 256, SMEM_BYTES, stream>>>(
      zb, w1t, w2t, b1, b2, gamma, beta, Wo, bo, cnt, slot_tok, yslot);

  yfinal_kernel<<<N_TOK / 256, 256, 0, stream>>>(tk_e, tk_slot, tk_w, yslot, out_y);
  stats_kernel<<<1, 256, 0, stream>>>(out_probs, cnt, out_imp, out_load);
}
